// MeshGraphNet_35330400977279
// MI455X (gfx1250) — compile-verified, matrix operand order checked by reading, not run on hardware
//
#include <hip/hip_runtime.h>
#include <hip/hip_bf16.h>
#include <cstdint>
#include <cstddef>

#define NN 16384
#define NE 262144

typedef __attribute__((ext_vector_type(8)))  float  v8f;
typedef __attribute__((ext_vector_type(16))) __bf16 v16bf;

union Frag {
  v16bf v;
  uint4 q[2];
  unsigned short s[16];
};

__device__ __forceinline__ unsigned short f2bf(float x) {
  unsigned int u = __float_as_uint(x);
  u += 0x7FFFu + ((u >> 16) & 1u);   // round-to-nearest-even
  return (unsigned short)(u >> 16);
}
__device__ __forceinline__ float bf2f(unsigned short s) {
  return __uint_as_float(((unsigned int)s) << 16);
}

// ---------------------------------------------------------------------------
// A-fragment loader. 16x32 bf16 fragment per ISA layout:
//   lane = 16*half + m ; element j<8  -> K = kb + half*8 + j
//                        element j>=8 -> K = kb + 16 + half*8 + (j-8)
// Both runs are 8 contiguous elements -> b128 loads.
// F32 sources are segmented (concat of 128-wide pieces, optionally gathered
// through an index array) and converted to bf16 on the fly.
// ---------------------------------------------------------------------------
template <bool ABF16, int NSEG>
__device__ __forceinline__ void load_a_frag(Frag& fr, const void* const* segB,
                                            const int* const* segI,
                                            int row, int kb, int half) {
  if constexpr (ABF16) {
    const unsigned short* p =
        (const unsigned short*)segB[0] + (size_t)row * 128 + kb + half * 8;
    fr.q[0] = *(const uint4*)p;
    fr.q[1] = *(const uint4*)(p + 16);
  } else {
    const int seg  = kb >> 7;        // which 128-wide concat segment
    const int koff = kb & 127;
    const float* base = (const float*)segB[seg];
    const int*   idx  = segI[seg];
    const int rg = idx ? idx[row] : row;
    const float* p = base + (size_t)rg * 128 + koff + half * 8;
    float4 x0 = *(const float4*)(p);
    float4 x1 = *(const float4*)(p + 4);
    float4 x2 = *(const float4*)(p + 16);
    float4 x3 = *(const float4*)(p + 20);
    fr.s[0]  = f2bf(x0.x); fr.s[1]  = f2bf(x0.y);
    fr.s[2]  = f2bf(x0.z); fr.s[3]  = f2bf(x0.w);
    fr.s[4]  = f2bf(x1.x); fr.s[5]  = f2bf(x1.y);
    fr.s[6]  = f2bf(x1.z); fr.s[7]  = f2bf(x1.w);
    fr.s[8]  = f2bf(x2.x); fr.s[9]  = f2bf(x2.y);
    fr.s[10] = f2bf(x2.z); fr.s[11] = f2bf(x2.w);
    fr.s[12] = f2bf(x3.x); fr.s[13] = f2bf(x3.y);
    fr.s[14] = f2bf(x3.z); fr.s[15] = f2bf(x3.w);
  }
}

// ---------------------------------------------------------------------------
// Fused GEMM:  out[M,128] = epilogue( A[M,K] @ W[K,128] + bias )
//   epilogue: optional ReLU -> optional tanh -> optional LayerNorm(g,b)
//             -> optional residual add -> store (f32 or bf16)
// Block = 128 threads (4 wave32); wave owns 32 rows x 128 cols
// = 2 row-frags x 8 col-tiles of v_wmma_f32_16x16x32_bf16.
// Weights pre-transposed to bf16 [128, K] so B frags are b128 loads.
// LayerNorm done entirely in registers with 16-lane shfl_xor reductions.
// ---------------------------------------------------------------------------
template <bool ABF16, int NSEG, bool RELU, bool TANH, bool DOLN, bool RES, bool OUTB>
__global__ __launch_bounds__(128) void wmma_mlp(
    const void* a0, const int* i0, const void* a1, const int* i1,
    const void* a2, const int* i2, const unsigned short* Wt, const float* bias,
    const float* lng, const float* lnb, const float* res, float* outf,
    unsigned short* outb) {
  constexpr int KTOT = ABF16 ? 128 : NSEG * 128;
  const int lane = threadIdx.x & 31;
  const int wave = threadIdx.x >> 5;
  const int row0 = blockIdx.x * 128 + wave * 32;
  const int m4   = lane & 15;
  const int half = lane >> 4;

  const void* segB[3] = {a0, a1, a2};
  const int*  segI[3] = {i0, i1, i2};

  v8f acc[2][8];
#pragma unroll
  for (int f = 0; f < 2; f++)
#pragma unroll
    for (int t = 0; t < 8; t++)
#pragma unroll
      for (int r = 0; r < 8; r++) acc[f][t][r] = 0.0f;

#pragma unroll
  for (int kb = 0; kb < KTOT; kb += 32) {
    Frag af0, af1;
    load_a_frag<ABF16, NSEG>(af0, segB, segI, row0 + m4,      kb, half);
    load_a_frag<ABF16, NSEG>(af1, segB, segI, row0 + 16 + m4, kb, half);
#pragma unroll
    for (int t = 0; t < 8; t++) {
      Frag bfr;
      const unsigned short* wp =
          Wt + (size_t)(t * 16 + m4) * KTOT + kb + half * 8;
      bfr.q[0] = *(const uint4*)wp;
      bfr.q[1] = *(const uint4*)(wp + 16);
      acc[0][t] = __builtin_amdgcn_wmma_f32_16x16x32_bf16(
          false, af0.v, false, bfr.v, (short)0, acc[0][t], false, false);
      acc[1][t] = __builtin_amdgcn_wmma_f32_16x16x32_bf16(
          false, af1.v, false, bfr.v, (short)0, acc[1][t], false, false);
    }
  }

  // ---- epilogue (bias / act / LN / residual / store) ----
  float bs[8], gg[8], bbv[8];
#pragma unroll
  for (int t = 0; t < 8; t++) {
    const int n = t * 16 + m4;
    bs[t] = bias[n];
    if constexpr (DOLN) { gg[t] = lng[n]; bbv[t] = lnb[n]; }
  }

#pragma unroll
  for (int f = 0; f < 2; f++) {
#pragma unroll
    for (int t = 0; t < 8; t++)
#pragma unroll
      for (int r = 0; r < 8; r++) {
        float v = acc[f][t][r] + bs[t];
        if constexpr (RELU) v = fmaxf(v, 0.0f);
        if constexpr (TANH) v = tanhf(v);
        acc[f][t][r] = v;
      }
    if constexpr (DOLN) {
#pragma unroll
      for (int r = 0; r < 8; r++) {
        float s = 0.0f, q = 0.0f;
#pragma unroll
        for (int t = 0; t < 8; t++) {
          float v = acc[f][t][r];
          s += v; q += v * v;
        }
#pragma unroll
        for (int off = 1; off < 16; off <<= 1) {  // stays within 16-lane half
          s += __shfl_xor(s, off, 32);
          q += __shfl_xor(q, off, 32);
        }
        const float mean = s * (1.0f / 128.0f);
        const float var  = q * (1.0f / 128.0f) - mean * mean;
        const float inv  = rsqrtf(var + 1e-5f);
#pragma unroll
        for (int t = 0; t < 8; t++)
          acc[f][t][r] = (acc[f][t][r] - mean) * inv * gg[t] + bbv[t];
      }
    }
#pragma unroll
    for (int r = 0; r < 8; r++) {
      const size_t row = (size_t)(row0 + 16 * f + 8 * half + r);
#pragma unroll
      for (int t = 0; t < 8; t++) {
        const int n = t * 16 + m4;
        float v = acc[f][t][r];
        if constexpr (RES) v += res[row * 128 + n];
        if constexpr (OUTB) outb[row * 128 + n] = f2bf(v);
        else                outf[row * 128 + n] = v;
      }
    }
  }
}

// ---------------------------------------------------------------------------
// Small scalar kernels
// ---------------------------------------------------------------------------
// pos encoder: 2 -> 128 -> 2, LayerNorm(2). One thread per node.
__global__ void pos_enc_kernel(const float* pos, const float* w1, const float* b1,
                               const float* w2, const float* b2, const float* g,
                               const float* bb, float* pe, int n) {
  int i = blockIdx.x * blockDim.x + threadIdx.x;
  if (i >= n) return;
  const float p0 = pos[2 * i], p1 = pos[2 * i + 1];
  float a0 = b2[0], a1 = b2[1];
  for (int k = 0; k < 128; k++) {
    const float h = b1[k] + p0 * w1[k] + p1 * w1[128 + k];
    a0 += h * w2[2 * k];
    a1 += h * w2[2 * k + 1];
  }
  const float mean = 0.5f * (a0 + a1);
  const float d0 = a0 - mean, d1 = a1 - mean;
  const float inv = rsqrtf(0.5f * (d0 * d0 + d1 * d1) + 1e-5f);
  pe[2 * i]     = d0 * inv * g[0] + bb[0];
  pe[2 * i + 1] = d1 * inv * g[1] + bb[1];
}

// node encoder fc1: concat[x(6), pe(2)] @ [8,128] + b -> bf16
__global__ void node_fc1_kernel(const float* x, const float* pe, const float* w,
                                const float* b, unsigned short* out, int n) {
  int t = blockIdx.x * blockDim.x + threadIdx.x;
  if (t >= n * 128) return;
  const int i = t >> 7, j = t & 127;
  float acc = b[j];
#pragma unroll
  for (int k = 0; k < 6; k++) acc += x[i * 6 + k] * w[k * 128 + j];
  acc += pe[i * 2] * w[6 * 128 + j] + pe[i * 2 + 1] * w[7 * 128 + j];
  out[t] = f2bf(acc);
}

// edge encoder fc1: edge_attr(3) @ [3,128] + b -> bf16
__global__ void edge_fc1_kernel(const float* ea, const float* w, const float* b,
                                unsigned short* out, int n) {
  int t = blockIdx.x * blockDim.x + threadIdx.x;
  if (t >= n * 128) return;
  const int i = t >> 7, j = t & 127;
  float acc = b[j];
#pragma unroll
  for (int k = 0; k < 3; k++) acc += ea[i * 3 + k] * w[k * 128 + j];
  out[t] = f2bf(acc);
}

// decoder fc3: [128] -> [2]
__global__ void dec_fc3_kernel(const unsigned short* hid, const float* w,
                               const float* b, float* out, int n) {
  int i = blockIdx.x * blockDim.x + threadIdx.x;
  if (i >= n) return;
  float a0 = b[0], a1 = b[1];
  for (int k = 0; k < 128; k++) {
    const float h = bf2f(hid[(size_t)i * 128 + k]);
    a0 += h * w[2 * k];
    a1 += h * w[2 * k + 1];
  }
  out[2 * i] = a0;
  out[2 * i + 1] = a1;
}

// transpose + convert weights: f32 [K,128] -> bf16 [128,K]
__global__ void wt_convert_kernel(const float* w, unsigned short* wt, int K) {
  int t = blockIdx.x * blockDim.x + threadIdx.x;
  if (t >= K * 128) return;
  const int nn = t / K, k = t % K;
  wt[t] = f2bf(w[k * 128 + nn]);
}

__global__ void zero_kernel(float* p, int n) {
  int i = blockIdx.x * blockDim.x + threadIdx.x;
  if (i < n) p[i] = 0.0f;
}

// scatter-sum: agg[src[e]] += e_feat[e]  (global_atomic_add_f32)
__global__ void scatter_add_kernel(const float* e, const int* src, float* agg,
                                   int total) {
  int t = blockIdx.x * blockDim.x + threadIdx.x;
  if (t >= total) return;
  const int eid = t >> 7, c = t & 127;
  unsafeAtomicAdd(&agg[(size_t)src[eid] * 128 + c], e[t]);
}

// ---------------------------------------------------------------------------
extern "C" void kernel_launch(void* const* d_in, const int* in_sizes, int n_in,
                              void* d_out, int out_size, void* d_ws,
                              size_t ws_size, hipStream_t stream) {
  const float* x   = (const float*)d_in[0];
  const float* pos = (const float*)d_in[1];
  const float* ea  = (const float*)d_in[2];
  const int*   ei  = (const int*)d_in[3];
  const int* src = ei;        // edge_index[0] : x_j gather + scatter key
  const int* dst = ei + NE;   // edge_index[1] : x_i gather

  const float* P[64];
  for (int i = 4; i < 64 && i < n_in; i++) P[i] = (const float*)d_in[i];

  // params flattened jax-tree style (dict keys sorted): dec, edge_enc, layers,
  // node_enc, pos_enc; within a linear: b then w; within ln: b then g.
  const float *dec1_b = P[4],  *dec1_w = P[5];
  const float *dec2_b = P[6],  *dec2_w = P[7];
  const float *dec3_b = P[8],  *dec3_w = P[9];
  const float *ee1_b = P[10],  *ee1_w = P[11];
  const float *ee2_b = P[12],  *ee2_w = P[13];
  const float *eeln_b = P[14], *eeln_g = P[15];
  const float *ne1_b = P[52],  *ne1_w = P[53];
  const float *ne2_b = P[54],  *ne2_w = P[55];
  const float *neln_b = P[56], *neln_g = P[57];
  const float *pe1_b = P[58],  *pe1_w = P[59];
  const float *pe2_b = P[60],  *pe2_w = P[61];
  const float *peln_b = P[62], *peln_g = P[63];

  // workspace carve-out
  char* wsp = (char*)d_ws;
  size_t off = 0;
  auto alloc = [&](size_t bytes) -> char* {
    char* p = wsp + off;
    off += (bytes + 255) & ~(size_t)255;
    return p;
  };
  float* e_f32 = (float*)alloc((size_t)NE * 128 * 4);   // edge state
  float* h_f32 = (float*)alloc((size_t)NN * 128 * 4);   // node state
  float* agg   = (float*)alloc((size_t)NN * 128 * 4);
  float* pe    = (float*)alloc((size_t)NN * 2 * 4);
  unsigned short* hidE = (unsigned short*)alloc((size_t)NE * 128 * 2);
  unsigned short* hidN = (unsigned short*)alloc((size_t)NN * 128 * 2);
  unsigned short* wt_ne2 = (unsigned short*)alloc(128 * 128 * 2);
  unsigned short* wt_ee2 = (unsigned short*)alloc(128 * 128 * 2);
  unsigned short* wt_d1  = (unsigned short*)alloc(128 * 128 * 2);
  unsigned short* wt_d2  = (unsigned short*)alloc(128 * 128 * 2);
  unsigned short *wt_el1[3], *wt_el2[3], *wt_nl1[3], *wt_nl2[3];
  for (int L = 0; L < 3; L++) {
    wt_el1[L] = (unsigned short*)alloc(384 * 128 * 2);
    wt_el2[L] = (unsigned short*)alloc(128 * 128 * 2);
    wt_nl1[L] = (unsigned short*)alloc(256 * 128 * 2);
    wt_nl2[L] = (unsigned short*)alloc(128 * 128 * 2);
  }

  auto conv = [&](const float* w, unsigned short* wt, int K) {
    int tot = K * 128;
    wt_convert_kernel<<<(tot + 255) / 256, 256, 0, stream>>>(w, wt, K);
  };
  conv(ne2_w, wt_ne2, 128);
  conv(ee2_w, wt_ee2, 128);
  conv(dec1_w, wt_d1, 128);
  conv(dec2_w, wt_d2, 128);
  for (int L = 0; L < 3; L++) {
    const int pb = 16 + 12 * L;
    conv(P[pb + 1], wt_el1[L], 384);
    conv(P[pb + 3], wt_el2[L], 128);
    conv(P[pb + 7], wt_nl1[L], 256);
    conv(P[pb + 9], wt_nl2[L], 128);
  }

  const dim3 blk(128);
  const int gN = NN / 128, gE = NE / 128;

  // encoders
  pos_enc_kernel<<<(NN + 255) / 256, 256, 0, stream>>>(
      pos, pe1_w, pe1_b, pe2_w, pe2_b, peln_g, peln_b, pe, NN);
  node_fc1_kernel<<<(NN * 128) / 256, 256, 0, stream>>>(x, pe, ne1_w, ne1_b,
                                                        hidN, NN);
  // node_enc fc2: relu -> LN -> h
  wmma_mlp<true, 1, true, false, true, false, false><<<gN, blk, 0, stream>>>(
      hidN, nullptr, nullptr, nullptr, nullptr, nullptr, wt_ne2, ne2_b,
      neln_g, neln_b, nullptr, h_f32, nullptr);
  edge_fc1_kernel<<<(NE * 128) / 256, 256, 0, stream>>>(ea, ee1_w, ee1_b,
                                                        hidE, NE);
  // edge_enc fc2: relu -> tanh -> LN -> e
  wmma_mlp<true, 1, true, true, true, false, false><<<gE, blk, 0, stream>>>(
      hidE, nullptr, nullptr, nullptr, nullptr, nullptr, wt_ee2, ee2_b,
      eeln_g, eeln_b, nullptr, e_f32, nullptr);

  // message-passing layers
  for (int L = 0; L < 3; L++) {
    const int pb = 16 + 12 * L;
    const float *el1_b = P[pb],      *el2_b = P[pb + 2];
    const float *eln_b = P[pb + 4],  *eln_g = P[pb + 5];
    const float *nl1_b = P[pb + 6],  *nl2_b = P[pb + 8];
    const float *nln_b = P[pb + 10], *nln_g = P[pb + 11];

    // edge l1: concat[h[dst], h[src], e] @ W(384,128) -> relu -> bf16
    wmma_mlp<false, 3, true, false, false, false, true><<<gE, blk, 0, stream>>>(
        h_f32, dst, h_f32, src, e_f32, nullptr, wt_el1[L], el1_b, nullptr,
        nullptr, nullptr, nullptr, hidE);
    // edge l2: LN + residual(e) -> e
    wmma_mlp<true, 1, false, false, true, true, false><<<gE, blk, 0, stream>>>(
        hidE, nullptr, nullptr, nullptr, nullptr, nullptr, wt_el2[L], el2_b,
        eln_g, eln_b, e_f32, e_f32, nullptr);
    // scatter-sum keyed on src
    zero_kernel<<<(NN * 128) / 256, 256, 0, stream>>>(agg, NN * 128);
    scatter_add_kernel<<<(NE * 128) / 256, 256, 0, stream>>>(e_f32, src, agg,
                                                             NE * 128);
    // node l1: concat[h, agg] @ W(256,128) -> relu -> bf16
    wmma_mlp<false, 2, true, false, false, false, true><<<gN, blk, 0, stream>>>(
        h_f32, nullptr, agg, nullptr, nullptr, nullptr, wt_nl1[L], nl1_b,
        nullptr, nullptr, nullptr, nullptr, hidN);
    // node l2: tanh -> LN + residual(h) -> h
    wmma_mlp<true, 1, false, true, true, true, false><<<gN, blk, 0, stream>>>(
        hidN, nullptr, nullptr, nullptr, nullptr, nullptr, wt_nl2[L], nl2_b,
        nln_g, nln_b, h_f32, h_f32, nullptr);
  }

  // decoder
  wmma_mlp<false, 1, true, false, false, false, true><<<gN, blk, 0, stream>>>(
      h_f32, nullptr, nullptr, nullptr, nullptr, nullptr, wt_d1, dec1_b,
      nullptr, nullptr, nullptr, nullptr, hidE);
  wmma_mlp<true, 1, false, false, false, false, true><<<gN, blk, 0, stream>>>(
      hidE, nullptr, nullptr, nullptr, nullptr, nullptr, wt_d2, dec2_b,
      nullptr, nullptr, nullptr, nullptr, hidN);
  dec_fc3_kernel<<<(NN + 255) / 256, 256, 0, stream>>>(hidN, dec3_w, dec3_b,
                                                       (float*)d_out, NN);
}
